// TritonLiteMLAFwd_6519760355475
// MI455X (gfx1250) — matrix operand order repaired and checked by reading, not compiled, not run on hardware
//
#include <hip/hip_runtime.h>

// ---------- config ----------
// Prefer TDM (TENSOR_LOAD_TO_LDS) for GEMM staging; fall back to per-lane
// global_load_async_to_lds_b128; both tracked by their CDNA5 counters.
#if defined(__has_builtin)
#if __has_builtin(__builtin_amdgcn_tensor_load_to_lds)
#define USE_TDM 1
#endif
#endif
#ifndef USE_TDM
#define USE_TDM 0
#endif

// ---------- types ----------
typedef __attribute__((ext_vector_type(16))) __bf16          v16bf;
typedef __attribute__((ext_vector_type(8)))  float           v8f;
typedef __attribute__((ext_vector_type(16))) unsigned short  v16u;
typedef __attribute__((ext_vector_type(8)))  unsigned short  v8u;
typedef __attribute__((ext_vector_type(4)))  unsigned int    u32x4;
typedef __attribute__((ext_vector_type(8)))  int             i32x8;
typedef __attribute__((ext_vector_type(4)))  int             i32x4;

constexpr int LS = 72; // GEMM LDS row stride in halfs (144 B: 16B-aligned, bank-spread)

// ---------- helpers ----------
static __device__ __forceinline__ unsigned short f2bf(float f) {
    unsigned int u = __float_as_uint(f);
    u += 0x7FFFu + ((u >> 16) & 1u);     // round-to-nearest-even
    return (unsigned short)(u >> 16);
}
static __device__ __forceinline__ float bf2f(unsigned short h) {
    return __uint_as_float(((unsigned int)h) << 16);
}
static __device__ __forceinline__ v16u cat16(v8u lo, v8u hi) {
    v16u r;
#pragma unroll
    for (int i = 0; i < 8; i++) { r[i] = lo[i]; r[i + 8] = hi[i]; }
    return r;
}
// D(f32 16x16) = A(bf16 16x32) * B(bf16 32x16) + C
static __device__ __forceinline__ v8f wmma_bf16(v16u a, v16u b, v8f c) {
    return __builtin_amdgcn_wmma_f32_16x16x32_bf16(
        false, __builtin_bit_cast(v16bf, a),
        false, __builtin_bit_cast(v16bf, b),
        (short)0, c, false, false);
}
static __device__ __forceinline__ unsigned int lds_addr32(const void* p) {
    // LDS aperture: flat addr low 32 bits == wave-relative LDS byte address
    return (unsigned int)(unsigned long long)p;
}
static __device__ __forceinline__ void wait_tensor0() {
#if defined(__has_builtin) && __has_builtin(__builtin_amdgcn_s_wait_tensorcnt)
    __builtin_amdgcn_s_wait_tensorcnt(0);
#else
    asm volatile("s_wait_tensorcnt 0x0" ::: "memory");
#endif
}

#if USE_TDM
// ---------- TDM: one descriptor moves a whole 128x64-half tile ----------
// 2D tile: 128 rows x 128 B, global row stride = ldg halfs; HW pads each
// LDS row by 4 DWORDs after every 32 DWORDs -> 72-half LDS row stride.
static __device__ __forceinline__ void tdm_load_tile(unsigned int lds,
                                                     const unsigned short* g,
                                                     int ldg) {
    const unsigned long long ga = (unsigned long long)g;
    u32x4 g0;
    g0[0] = 1u;                                      // count=1, user desc
    g0[1] = lds;                                     // lds_addr (bytes)
    g0[2] = (unsigned int)ga;                        // global_addr[31:0]
    g0[3] = (unsigned int)(ga >> 32) | (2u << 30);   // global_addr[56:32] | type=2
    i32x8 g1;
    g1[0] = (1 << 16)    // data_size = 2 bytes
          | (1 << 20)    // pad_enable
          | (4 << 22)    // pad_interval: 32 DWORDs (64 halfs)
          | (3 << 25);   // pad_amount: 4 DWORDs (8 halfs)
    g1[1] = 64 << 16;    // tensor_dim0 = 64 (lo 16)
    g1[2] = 128 << 16;   // tensor_dim0 hi=0 | tensor_dim1 = 128 (lo 16)
    g1[3] = 64 << 16;    // tensor_dim1 hi=0 | tile_dim0 = 64
    g1[4] = 128;         // tile_dim1 = 128 | tile_dim2 = 0
    g1[5] = ldg;         // tensor_dim0_stride[31:0] (halfs)
    g1[6] = 0;
    g1[7] = 0;
    i32x4 z4; z4[0] = z4[1] = z4[2] = z4[3] = 0;
    i32x8 z8;
#pragma unroll
    for (int i = 0; i < 8; i++) z8[i] = 0;
    // 6-arg toolchain variant: (g0, g1, g2, g3, ext, cpol)
    __builtin_amdgcn_tensor_load_to_lds(g0, g1, z4, z4, z8, 0);
}
#else
// ---------- fallback: per-lane async global->LDS copies (ASYNCcnt) ----------
static __device__ __forceinline__ void cp16_g2l(unsigned short* lds_dst,
                                               const unsigned short* g_src) {
    asm volatile("global_load_async_to_lds_b128 %0, %1, off"
                 :: "v"(lds_addr32(lds_dst)), "v"((unsigned long long)g_src)
                 : "memory");
}
static __device__ __forceinline__ void async_wait0() {
    asm volatile("s_wait_asynccnt 0x0" ::: "memory");
}
static __device__ __forceinline__ void stage_tile(unsigned short* dst,
                                                  const unsigned short* src,
                                                  long ldg, int tid) {
    const int srow = tid >> 1;          // 128 rows, 2 threads/row
    const int scol = (tid & 1) * 32;    // 32 halfs per thread
    const unsigned short* s = src + (long)srow * ldg + scol;
    unsigned short* d = dst + srow * LS + scol;
#pragma unroll
    for (int i = 0; i < 4; i++) cp16_g2l(d + 8 * i, s + 8 * i);
}
#endif

// =====================================================================
// K1: f32 -> bf16 prepack (one pass; kills per-tile conversion in GEMMs)
// =====================================================================
__global__ __launch_bounds__(256) void pack_bf16(const float* __restrict__ src,
                                                 unsigned short* __restrict__ dst,
                                                 int n8) {
    const int i = blockIdx.x * blockDim.x + threadIdx.x;
    if (i >= n8) return;
    const float4* s = (const float4*)src + 2 * (long)i;
    const float4 a = s[0], b = s[1];
    unsigned short t[8] = { f2bf(a.x), f2bf(a.y), f2bf(a.z), f2bf(a.w),
                            f2bf(b.x), f2bf(b.y), f2bf(b.z), f2bf(b.w) };
    *(v8u*)(dst + 8 * (long)i) = *(const v8u*)t;
}

// =====================================================================
// GEMM (NT): Out[m,n] = sum_k A[m,k]*W[n,k]; A,W bf16 K-fast.
// 256 thr = 8 waves (4x2), tile 128x128x64, double-buffered TDM staging.
// OUT_BF16: ReLU on cols<relu_cols, bf16 store. else: +bias, f32 store.
// =====================================================================
template <bool OUT_BF16>
__global__ __launch_bounds__(256) void gemm_nt(
    const unsigned short* __restrict__ A, const unsigned short* __restrict__ W,
    void* __restrict__ Outv, const float* __restrict__ bias,
    int M, int N, int K, int relu_cols)
{
    constexpr int BM = 128, BN = 128, BK = 64;
    __shared__ __align__(16) unsigned short sA[2][BM * LS];
    __shared__ __align__(16) unsigned short sB[2][BN * LS];

    const int tid  = threadIdx.x;
    const int lane = tid & 31, w = tid >> 5;
    const int wm = w & 3, wn = w >> 2;        // wave grid 4(M) x 2(N)
    const int r = lane & 15, hf = lane >> 4;
    const long m0 = (long)blockIdx.x * BM;
    const long n0 = (long)blockIdx.y * BN;
    const unsigned short* Ab = A + m0 * K;
    const unsigned short* Wb = W + n0 * K;

    v8f acc[2][4];
#pragma unroll
    for (int a = 0; a < 2; a++)
#pragma unroll
        for (int b = 0; b < 4; b++)
#pragma unroll
            for (int e = 0; e < 8; e++) acc[a][b][e] = 0.f;

    // prologue: stage k-tile 0
#if USE_TDM
    if (tid < 32) {   // one wave drives the tensor DMA engine
        tdm_load_tile(lds_addr32(&sA[0][0]), Ab, K);
        tdm_load_tile(lds_addr32(&sB[0][0]), Wb, K);
        wait_tensor0();
    }
#else
    stage_tile(sA[0], Ab, K, tid);
    stage_tile(sB[0], Wb, K, tid);
    async_wait0();
#endif
    __syncthreads();

    int cur = 0;
    for (int k0 = 0; k0 < K; k0 += BK) {
        // issue staging of next k-tile into the other buffer
        if (k0 + BK < K) {
#if USE_TDM
            if (tid < 32) {
                tdm_load_tile(lds_addr32(&sA[cur ^ 1][0]), Ab + k0 + BK, K);
                tdm_load_tile(lds_addr32(&sB[cur ^ 1][0]), Wb + k0 + BK, K);
            }
#else
            stage_tile(sA[cur ^ 1], Ab + k0 + BK, K, tid);
            stage_tile(sB[cur ^ 1], Wb + k0 + BK, K, tid);
#endif
        }
        // compute on current buffer
#pragma unroll
        for (int kk = 0; kk < BK; kk += 32) {
            v16u areg[2], breg[4];
#pragma unroll
            for (int ms = 0; ms < 2; ms++) {
                // A bf16 16x32: lane=row r, chunks k=8h..+8 and 16+8h..+8
                const unsigned short* s =
                    &sA[cur][(wm * 32 + ms * 16 + r) * LS + kk + 8 * hf];
                areg[ms] = cat16(*(const v8u*)s, *(const v8u*)(s + 16));
            }
#pragma unroll
            for (int ns = 0; ns < 4; ns++) {
                // B bf16 32x16: lane=col r, chunks k=8h..+8 and 16+8h..+8
                const unsigned short* s =
                    &sB[cur][(wn * 64 + ns * 16 + r) * LS + kk + 8 * hf];
                breg[ns] = cat16(*(const v8u*)s, *(const v8u*)(s + 16));
            }
#pragma unroll
            for (int ms = 0; ms < 2; ms++)
#pragma unroll
                for (int ns = 0; ns < 4; ns++)
                    acc[ms][ns] = wmma_bf16(areg[ms], breg[ns], acc[ms][ns]);
        }
#if USE_TDM
        if (tid < 32) wait_tensor0();
#else
        async_wait0();
#endif
        __syncthreads();
        cur ^= 1;
    }

    // epilogue: C/D element (m = 8*hf + j, n = r) per 16x16 tile
#pragma unroll
    for (int ms = 0; ms < 2; ms++) {
#pragma unroll
        for (int ns = 0; ns < 4; ns++) {
            const long col = n0 + wn * 64 + ns * 16 + r;
#pragma unroll
            for (int j = 0; j < 8; j++) {
                const long row = m0 + wm * 32 + ms * 16 + 8 * hf + j;
                float v = acc[ms][ns][j];
                if (OUT_BF16) {
                    if (col < relu_cols) v = v > 0.f ? v : 0.f;
                    ((unsigned short*)Outv)[row * (long)N + col] = f2bf(v);
                } else {
                    ((float*)Outv)[row * (long)N + col] = v + bias[col];
                }
            }
        }
    }
}

// =====================================================================
// vk kernel: vk[b,h,c,d] = sum_n v[b,n,h,c]*k[b,n,h,d]; row c=32 = sum_n k
// Split-K over tokens (256/block), fp32 atomic accumulation.
// =====================================================================
__global__ __launch_bounds__(128) void vk_accum(
    const unsigned short* __restrict__ qkv, float* __restrict__ vkbuf)
{
    constexpr int NT = 256, TS = 264;
    __shared__ __align__(16) unsigned short kT[32 * TS]; // [d][n]
    __shared__ __align__(16) unsigned short vT[32 * TS]; // [c][n]

    const int tid = threadIdx.x, lane = tid & 31, w = tid >> 5;
    const int bh = blockIdx.x;            // b*16 + h
    const int hh = bh & 15;
    const long nb = (long)(bh >> 4) * 16384 + (long)blockIdx.y * NT;
    const int r = lane & 15, hf = lane >> 4;

#pragma unroll
    for (int rep = 0; rep < 2; rep++) {
        const int nl = tid + 128 * rep;
        const long n = nb + nl;
        union { uint4 q[4]; unsigned short s[32]; } uk, uv;
        const uint4* kp = (const uint4*)(qkv + n * 1536 + 512 + 32 * hh);
        const uint4* vp = (const uint4*)(qkv + n * 1536 + 1024 + 32 * hh);
#pragma unroll
        for (int i = 0; i < 4; i++) { uk.q[i] = kp[i]; uv.q[i] = vp[i]; }
#pragma unroll
        for (int d = 0; d < 32; d++) {
            kT[d * TS + nl] = uk.s[d];
            vT[d * TS + nl] = uv.s[d];
        }
    }
    __syncthreads();

    { // normalizer row: vk[32][d] += sum_n k[n][d]
        float s = 0.f;
        const int base = w * 64;
#pragma unroll 8
        for (int i = 0; i < 64; i++) s += bf2f(kT[lane * TS + base + i]);
        atomicAdd(&vkbuf[(bh * 48 + 32) * 32 + lane], s);
    }

    v8f acc[2][2];
#pragma unroll
    for (int a = 0; a < 2; a++)
#pragma unroll
        for (int b = 0; b < 2; b++)
#pragma unroll
            for (int e = 0; e < 8; e++) acc[a][b][e] = 0.f;

#pragma unroll
    for (int kk = 0; kk < 64; kk += 32) {
        const int nbase = w * 64 + kk;
        v16u areg[2], breg[2];
#pragma unroll
        for (int mc = 0; mc < 2; mc++) { // A = v: M=c, K=n
            const unsigned short* s = &vT[(mc * 16 + r) * TS + nbase + 8 * hf];
            areg[mc] = cat16(*(const v8u*)s, *(const v8u*)(s + 16));
        }
#pragma unroll
        for (int nc = 0; nc < 2; nc++) { // B = k: K=n, N=d
            const unsigned short* s = &kT[(nc * 16 + r) * TS + nbase + 8 * hf];
            breg[nc] = cat16(*(const v8u*)s, *(const v8u*)(s + 16));
        }
#pragma unroll
        for (int mc = 0; mc < 2; mc++)
#pragma unroll
            for (int nc = 0; nc < 2; nc++)
                acc[mc][nc] = wmma_bf16(areg[mc], breg[nc], acc[mc][nc]);
    }

#pragma unroll
    for (int mc = 0; mc < 2; mc++)
#pragma unroll
        for (int nc = 0; nc < 2; nc++) {
            const int d = nc * 16 + r;
#pragma unroll
            for (int j = 0; j < 8; j++) {
                const int c = mc * 16 + 8 * hf + j;
                atomicAdd(&vkbuf[(bh * 48 + c) * 32 + d], acc[mc][nc][j]);
            }
        }
}

// =====================================================================
// vk_q + divide: per (b,h): [128 tokens x 32] @ vk' -> divide -> bf16
// =====================================================================
__global__ __launch_bounds__(128) void vkq_div(
    const unsigned short* __restrict__ qkv, const float* __restrict__ vkbuf,
    unsigned short* __restrict__ pin)
{
    __shared__ __align__(16) unsigned short vkb[48 * 32]; // [c][d] bf16
    const int tid = threadIdx.x, lane = tid & 31, w = tid >> 5;
    const int bh = blockIdx.x, hh = bh & 15;
    const long nb = (long)(bh >> 4) * 16384 + (long)blockIdx.y * 128;
    const int r = lane & 15, hf = lane >> 4;

    for (int i = tid; i < 48 * 32; i += 128) vkb[i] = f2bf(vkbuf[bh * 1536 + i]);
    __syncthreads();

    v16u breg[3];
#pragma unroll
    for (int nt = 0; nt < 3; nt++) { // B[k=d, col=c] = vk[c][d]
        const unsigned short* s = &vkb[(nt * 16 + r) * 32 + 8 * hf];
        breg[nt] = cat16(*(const v8u*)s, *(const v8u*)(s + 16));
    }

#pragma unroll
    for (int ms = 0; ms < 2; ms++) {
        const long n = nb + w * 32 + ms * 16 + r;
        const unsigned short* qp = qkv + n * 1536 + 32 * hh; // q slot, d-fast
        v16u a = cat16(*(const v8u*)(qp + 8 * hf), *(const v8u*)(qp + 16 + 8 * hf));

        v8f acc[3];
#pragma unroll
        for (int t = 0; t < 3; t++)
#pragma unroll
            for (int e = 0; e < 8; e++) acc[t][e] = 0.f;
#pragma unroll
        for (int t = 0; t < 3; t++) acc[t] = wmma_bf16(a, breg[t], acc[t]);

#pragma unroll
        for (int j = 0; j < 8; j++) {
            const float norm = __shfl(acc[2][j], lane & 16, 32) + 1e-15f;
            const long row = nb + w * 32 + ms * 16 + 8 * hf + j;
            unsigned short* op = pin + row * 512 + 32 * hh;
            op[r]      = f2bf(acc[0][j] / norm);
            op[16 + r] = f2bf(acc[1][j] / norm);
        }
    }
}

// =====================================================================
// launcher
// =====================================================================
extern "C" void kernel_launch(void* const* d_in, const int* in_sizes, int n_in,
                              void* d_out, int out_size, void* d_ws, size_t ws_size,
                              hipStream_t stream)
{
    const float* x  = (const float*)d_in[0];  // [4,16384,512]
    const float* wq = (const float*)d_in[1];  // [1536,512]
    const float* wp = (const float*)d_in[2];  // [512,512]
    const float* bp = (const float*)d_in[3];  // [512]

    char* ws = (char*)d_ws;
    unsigned short* qkv_b = (unsigned short*)ws;                      // 65536*1536 bf16
    float* vkbuf = (float*)(ws + 201326592);                          // 64*48*32 f32
    // x-bf16 aliases proj_input: x_b dies after GEMM1, pin born at K4
    unsigned short* xb  = (unsigned short*)(ws + 201719808);          // 65536*512 bf16
    unsigned short* pin = xb;
    unsigned short* wqb = (unsigned short*)(ws + 268828672);          // 1536*512 bf16
    unsigned short* wpb = (unsigned short*)(ws + 270401536);          // 512*512 bf16

    (void)hipMemsetAsync(vkbuf, 0, 64 * 48 * 32 * sizeof(float), stream);

    // K1: one-pass bf16 prepack of activations + weights
    pack_bf16<<<dim3(4194304 / 256), 256, 0, stream>>>(x,  xb,  4194304); // 33.5M elems
    pack_bf16<<<dim3(384),           256, 0, stream>>>(wq, wqb, 98304);   // 786K
    pack_bf16<<<dim3(128),           256, 0, stream>>>(wp, wpb, 32768);   // 262K

    // K2: qkv = x @ Wqkv^T, ReLU on cols < 1024 (q,k), bf16 out
    gemm_nt<true><<<dim3(512, 12), 256, 0, stream>>>(
        xb, wqb, qkv_b, nullptr, 65536, 1536, 512, 1024);

    // K3: vk accumulation (split-K over tokens, f32 atomics)
    vk_accum<<<dim3(64, 64), 128, 0, stream>>>(qkv_b, vkbuf);

    // K4: q @ vk^T, divide by normalizer, bf16 out (overwrites xb region)
    vkq_div<<<dim3(64, 128), 128, 0, stream>>>(qkv_b, vkbuf, pin);

    // K5: out = proj_in @ Wproj^T + bias, f32 out
    gemm_nt<false><<<dim3(512, 4), 256, 0, stream>>>(
        pin, wpb, d_out, bp, 65536, 512, 512, 0);
}
